// PolyCntSketch_2903397892979
// MI455X (gfx1250) — matrix operand order repaired and checked by reading, not compiled, not run on hardware
//
#include <hip/hip_runtime.h>
#include <hip/hip_bf16.h>

// ---------------------------------------------------------------------------
// TensorSketch (CountSketch x3 + DFT product + inverse DFT) for MI455X gfx1250
//   B=8192, F=4096, N=512, DEGREE=3, GAMMA=1
// Strategy: batch 16 rows per block; DFT(512) as [16x512]x[512x512] matmuls
// on v_wmma_f32_16x16x32_bf16. Twiddle matrices (symmetric, shared by fwd &
// inv) pre-baked into d_ws in B-fragment-ready layout.
// ---------------------------------------------------------------------------

typedef __attribute__((ext_vector_type(16))) __bf16 v16bf;
typedef __attribute__((ext_vector_type(8)))  __bf16 v8bf;
typedef __attribute__((ext_vector_type(8)))  float  v8f;

#define NB   8192
#define NF   4096
#define NC   512
#define ROWS 16           // batch rows per block (WMMA M)
#define LSTR 520          // LDS row stride in bf16 halves (bank spread)

// LDS layout (dynamic):
//   [0,          98304)  : f32 count-sketch acc [3][16][512]   (phase 1)
//   [0,          16640)  : PR bf16 [16][520]                   (phase 3, aliases acc)
//   [16640,      33280)  : PI bf16 [16][520]
//   [98304,     148224)  : S  bf16 [3][16][520]                (phases 2-3)
#define ACC_BYTES  (3 * ROWS * NC * 4)
#define SB_OFF     ACC_BYTES
#define SB_BYTES   (3 * ROWS * LSTR * 2)
#define SMEM_BYTES (ACC_BYTES + SB_BYTES)

// ---- twiddle fill: ws holds cos then sin matrices, fragment-ready ----------
// B-fragment layout (v_wmma_*_16x16x32 B = 32x16, per ISA doc):
//   lane<16  : h0..7 <-> K = 0..7,  h8..15 <-> K = 16..23, N = lane
//   lane>=16 : h0..7 <-> K = 8..15, h8..15 <-> K = 24..31, N = lane-16
// Frag chunk for (tile, step): 32 lanes x 16 halves (32B/lane, contiguous).
// Half index = ((tile*16+step)*32 + lane)*16 + h.
__global__ __launch_bounds__(256) void ts_twiddle_fill(__bf16* __restrict__ ws) {
    int idx = blockIdx.x * 256 + threadIdx.x;       // [0, 2*512*512)
    if (idx >= 2 * NC * NC) return;
    int m    = idx >> 18;                            // 0 = cos, 1 = sin
    int rem  = idx & 262143;
    int ts   = rem >> 9;                             // tile*16 + step
    int lane = (rem >> 4) & 31;
    int h    = rem & 15;
    int tile = ts >> 4, step = ts & 15;
    int a = step * 32 + ((lane & 16) >> 1) + h + (h & 8); // reduction index
    int b = tile * 16 + (lane & 15);                 // output index
    int mm = (a * b) & (NC - 1);
    float th = (float)mm * (6.28318530717958647692f / (float)NC);
    float v  = m ? sinf(th) : cosf(th);
    ws[idx] = (__bf16)v;
}

// ---- fragment helpers ------------------------------------------------------
// A-matrix 16x32 bf16 layout (per ISA doc):
//   lane<16 : h0..7 <-> K=kb+0..7,  h8..15 <-> K=kb+16..23, M = lane
//   lane>=16: h0..7 <-> K=kb+8..15, h8..15 <-> K=kb+24..31, M = lane-16
__device__ __forceinline__ v16bf load_a_frag(const __bf16* base, int kb, int lane) {
    int off = (lane & 15) * LSTR + kb + ((lane & 16) ? 8 : 0);
    v8bf lo = *(const v8bf*)(base + off);
    v8bf hi = *(const v8bf*)(base + off + 16);
    v16bf a;
#pragma unroll
    for (int j = 0; j < 8; ++j) { a[j] = lo[j]; a[j + 8] = hi[j]; }
    return a;
}

__device__ __forceinline__ v16bf load_b_frag(const v16bf* ws, int tile, int step, int lane) {
    return ws[(tile * 16 + step) * 32 + lane];
}

#define WMMA_BF16(A, Bm, C) \
    __builtin_amdgcn_wmma_f32_16x16x32_bf16(false, (A), false, (Bm), (short)0, (C), false, false)

// ---- fused main kernel -----------------------------------------------------
__global__ __launch_bounds__(256) void ts_main(const float* __restrict__ X,
                                               const int* __restrict__ idxh,
                                               const int* __restrict__ bith,
                                               const __bf16* __restrict__ twid,
                                               float* __restrict__ out) {
    extern __shared__ char smem[];
    float*  acc = (float*)smem;                       // [3][16][512] f32
    __bf16* Sb  = (__bf16*)(smem + SB_OFF);           // [3][16][520] bf16
    __bf16* PR  = (__bf16*)smem;                      // [16][520] bf16 (alias acc)
    __bf16* PI  = (__bf16*)(smem + ROWS * LSTR * 2);  // [16][520] bf16 (alias acc)

    const int tid  = threadIdx.x;
    const int lane = tid & 31;
    const int wave = tid >> 5;
    const int r0   = blockIdx.x * ROWS;

    // ---- phase 1: count sketch into f32 LDS via ds_add_f32 atomics ----
    for (int i = tid; i < 3 * ROWS * NC; i += 256) acc[i] = 0.0f;
    __syncthreads();

    for (int f = tid; f < NF; f += 256) {
        int i0 = idxh[f], i1 = idxh[NF + f], i2 = idxh[2 * NF + f];
        float s0 = (float)(2 * bith[f] - 1);
        float s1 = (float)(2 * bith[NF + f] - 1);
        float s2 = (float)(2 * bith[2 * NF + f] - 1);
        const float* xp = X + (size_t)r0 * NF + f;
#pragma unroll 4
        for (int r = 0; r < ROWS; ++r) {
            float x = xp[(size_t)r * NF];
            atomicAdd(&acc[            (r << 9) + i0], x * s0);
            atomicAdd(&acc[ (ROWS<<9) + (r << 9) + i1], x * s1);
            atomicAdd(&acc[(2*ROWS<<9) + (r << 9) + i2], x * s2);
        }
    }
    __syncthreads();

    // ---- phase 2: convert sketches f32 -> bf16 (stride-520 rows) ----
    for (int i = tid; i < 3 * ROWS * NC; i += 256) {
        int d = i >> 13, rem = i & 8191, r = rem >> 9, k = rem & 511;
        Sb[(d * ROWS + r) * LSTR + k] = (__bf16)acc[i];
    }
    __syncthreads();

    const v16bf* wc = (const v16bf*)twid;        // cos frags
    const v16bf* ws = wc + (NC * NC / 16);       // sin frags

    // ---- phase 3: forward DFT (3 degrees) + complex triple product ----
    // Track F_d = Re_d - i*I_d with I_d = sum s*sin (no negation needed).
#pragma unroll 1
    for (int t = 0; t < 4; ++t) {
        int tile = wave * 4 + t;                 // frequency tile (16 freqs)
        v8f cR0 = {}, cI0 = {}, cR1 = {}, cI1 = {}, cR2 = {}, cI2 = {};
#pragma unroll 1
        for (int step = 0; step < 16; ++step) {
            __builtin_prefetch((const void*)&wc[(tile * 16 + ((step + 1) & 15)) * 32 + lane], 0, 1);
            v16bf bc = load_b_frag(wc, tile, step, lane);
            v16bf bs = load_b_frag(ws, tile, step, lane);
            int kb = step * 32;
            v16bf a0 = load_a_frag(Sb,                 kb, lane);
            v16bf a1 = load_a_frag(Sb +     ROWS*LSTR, kb, lane);
            v16bf a2 = load_a_frag(Sb + 2 * ROWS*LSTR, kb, lane);
            cR0 = WMMA_BF16(a0, bc, cR0);  cI0 = WMMA_BF16(a0, bs, cI0);
            cR1 = WMMA_BF16(a1, bc, cR1);  cI1 = WMMA_BF16(a1, bs, cI1);
            cR2 = WMMA_BF16(a2, bc, cR2);  cI2 = WMMA_BF16(a2, bs, cI2);
        }
        // (R0-iI0)(R1-iI1)(R2-iI2) = Rf - i*If  (elementwise on C/D layout)
        v8f Rp = cR0 * cR1 - cI0 * cI1;
        v8f Ip = cR0 * cI1 + cI0 * cR1;
        v8f Rf = Rp * cR2 - Ip * cI2;
        v8f If = Rp * cI2 + Ip * cR2;
        int col = tile * 16 + (lane & 15);
#pragma unroll
        for (int i = 0; i < 8; ++i) {
            int row = (lane & 16) ? (i + 8) : i;  // C/D layout: M index
            PR[row * LSTR + col] = (__bf16)Rf[i];
            PI[row * LSTR + col] = (__bf16)If[i];
        }
    }
    __syncthreads();

    // ---- phase 4: inverse DFT: out = (Rf*C + If*S)/512 ----
#pragma unroll 1
    for (int t = 0; t < 4; ++t) {
        int tile = wave * 4 + t;                 // output-time tile (16 samples)
        v8f c = {};
#pragma unroll 1
        for (int step = 0; step < 16; ++step) {
            v16bf ar = load_a_frag(PR, step * 32, lane);
            v16bf ai = load_a_frag(PI, step * 32, lane);
            v16bf bc = load_b_frag(wc, tile, step, lane);
            v16bf bs = load_b_frag(ws, tile, step, lane);
            c = WMMA_BF16(ar, bc, c);
            c = WMMA_BF16(ai, bs, c);
        }
        int col = tile * 16 + (lane & 15);
#pragma unroll
        for (int i = 0; i < 8; ++i) {
            int row = (lane & 16) ? (i + 8) : i;
            out[(size_t)(r0 + row) * NC + col] = c[i] * (1.0f / (float)NC);
        }
    }
}

// ---------------------------------------------------------------------------
extern "C" void kernel_launch(void* const* d_in, const int* in_sizes, int n_in,
                              void* d_out, int out_size, void* d_ws, size_t ws_size,
                              hipStream_t stream) {
    const float* X    = (const float*)d_in[0];
    const int*   idxh = (const int*)d_in[1];
    const int*   bith = (const int*)d_in[2];
    float*       out  = (float*)d_out;
    __bf16*      tw   = (__bf16*)d_ws;           // 2*512*512 bf16 = 1 MiB

    (void)in_sizes; (void)n_in; (void)out_size; (void)ws_size;

    hipFuncSetAttribute((const void*)ts_main,
                        hipFuncAttributeMaxDynamicSharedMemorySize, SMEM_BYTES);

    ts_twiddle_fill<<<(2 * NC * NC + 255) / 256, 256, 0, stream>>>(tw);
    ts_main<<<NB / ROWS, 256, SMEM_BYTES, stream>>>(X, idxh, bith, tw, out);
}
